// Decoder_85005992722750
// MI455X (gfx1250) — matrix operand order repaired and checked by reading, not compiled, not run on hardware
//
#include <hip/hip_runtime.h>
#include <math.h>

// ---------------------------------------------------------------------------
// MI455X (gfx1250) show-attend-tell decoder.
// All GEMMs via v_wmma_f32_16x16x32_f16 (wave32, 16x16 tiles, f32 accum).
// Sizes: N=64 P=64 ENC=1536 T=32 V=10000 E=A=H=512
// ---------------------------------------------------------------------------

typedef __attribute__((ext_vector_type(16))) _Float16 v16h;
typedef __attribute__((ext_vector_type(8)))  _Float16 v8h;
typedef __attribute__((ext_vector_type(8)))  float    v8f;

union V16H { v16h v; v8h h2[2]; _Float16 e[16]; };
union V8F  { v8f v; float e[8]; };

__device__ __forceinline__ float sigmoidf_(float x) { return 1.0f / (1.0f + __expf(-x)); }

// A tile loader: A row-major (M,K) f16, lane holds row m=lane&15,
// halves 0..15 = K kb..kb+15 contiguous, kb=(lane>>4)*16.
__device__ __forceinline__ v16h load_a_tile(const _Float16* __restrict__ A, int lda,
                                            int m0, int k, int lane) {
  int mn = lane & 15;
  int kb = (lane >> 4) * 16;
  const _Float16* p = A + (size_t)(m0 + mn) * lda + k + kb;
  V16H a;
  a.h2[0] = *(const v8h*)(p);
  a.h2[1] = *(const v8h*)(p + 8);
  return a.v;
}

// B tile loader: weights stored (N,K) row-major f16 (i.e. B^T), lane holds
// column n=lane&15, K contiguous kb2..kb2+15, kb2=(lane>>4)*16. One 32B load.
__device__ __forceinline__ v16h load_b_tile(const _Float16* __restrict__ W, int ldw,
                                            int n0, int k, int lane) {
  int nn = lane & 15;
  int kb = (lane >> 4) * 16;
  return *(const v16h*)(W + (size_t)(n0 + nn) * ldw + k + kb);
}

__device__ __forceinline__ v8f wmma_step(v16h a, v16h b, v8f c) {
  return __builtin_amdgcn_wmma_f32_16x16x32_f16(false, a, false, b, (short)0, c,
                                                false, false);
}

// ---------------------------------------------------------------------------
// Generic GEMM: C(M,N) = A(M,K) @ W(N,K)^T + bias.  One wave = one 16x16 tile.
// grid.x = N/16, grid.y = M/16, block = 32.
// ---------------------------------------------------------------------------
__global__ __launch_bounds__(32) void gemm_nk(
    const _Float16* __restrict__ A, int lda,
    const _Float16* __restrict__ W, int ldw,
    const float* __restrict__ bias,
    float* __restrict__ C, int ldc, int K) {
  const int lane = threadIdx.x;
  const int n0 = blockIdx.x * 16, m0 = blockIdx.y * 16;
  v8f acc = {};
  for (int k = 0; k < K; k += 32) {
    v16h a = load_a_tile(A, lda, m0, k, lane);
    v16h b = load_b_tile(W, ldw, n0, k, lane);
    acc = wmma_step(a, b, acc);
  }
  const int mn = lane & 15, hi = lane >> 4;
  const float bv = bias ? bias[n0 + mn] : 0.0f;
  V8F r; r.v = acc;
#pragma unroll
  for (int i = 0; i < 8; ++i)
    C[(size_t)(m0 + i + 8 * hi) * ldc + n0 + mn] = r.e[i] + bv;
}

// ---------------------------------------------------------------------------
// Gates GEMM: G(64,2048) = X(64,2048)@W_ih^T + H(64,512)@W_hh^T + b_ih + b_hh
// grid.x = 128 (N tiles), grid.y = 4 (M tiles), block = 32.
// ---------------------------------------------------------------------------
__global__ __launch_bounds__(32) void gates_gemm(
    const _Float16* __restrict__ X,    // 64 x 2048
    const _Float16* __restrict__ Wih,  // 2048 x 2048 (N,K)
    const _Float16* __restrict__ Hp,   // 64 x 512
    const _Float16* __restrict__ Whh,  // 2048 x 512 (N,K)
    const float* __restrict__ b_ih, const float* __restrict__ b_hh,
    float* __restrict__ G) {
  const int lane = threadIdx.x;
  const int n0 = blockIdx.x * 16, m0 = blockIdx.y * 16;
  v8f acc = {};
  for (int k = 0; k < 2048; k += 32)
    acc = wmma_step(load_a_tile(X, 2048, m0, k, lane),
                    load_b_tile(Wih, 2048, n0, k, lane), acc);
  for (int k = 0; k < 512; k += 32)
    acc = wmma_step(load_a_tile(Hp, 512, m0, k, lane),
                    load_b_tile(Whh, 512, n0, k, lane), acc);
  const int mn = lane & 15, hi = lane >> 4;
  const float bv = b_ih[n0 + mn] + b_hh[n0 + mn];
  V8F r; r.v = acc;
#pragma unroll
  for (int i = 0; i < 8; ++i)
    G[(size_t)(m0 + i + 8 * hi) * 2048 + n0 + mn] = r.e[i] + bv;
}

// ---------------------------------------------------------------------------
// h0/c0: (64,98304) @ init_W(98304,512).  init_W is K-major fp32, so each
// 32x16 B tile is staged through LDS with fp32->f16 conversion. Split-K with
// fp32 atomics. One wave covers all 4 M tiles (B read once per k-step).
// grid.x = 32 (N tiles), grid.z = 48 k-chunks * 2 (Wh / Wc), block = 32.
// ---------------------------------------------------------------------------
__global__ __launch_bounds__(32) void init_state_gemm(
    const _Float16* __restrict__ feat,  // (64, 98304) f16
    const float* __restrict__ Wh, const float* __restrict__ Wc,
    float* __restrict__ h0c0) {         // [2][64][512] f32, pre-zeroed
  const int lane = threadIdx.x;
  const int n0 = blockIdx.x * 16;
  const int which = blockIdx.z & 1;
  const int chunk = blockIdx.z >> 1;
  const float* __restrict__ W = which ? Wc : Wh;
  float* __restrict__ C = h0c0 + which * (64 * 512);

  __shared__ _Float16 lbs[32 * 16];
  const int mn = lane & 15, hi = lane >> 4;
  const int kbB = hi * 16;

  v8f acc0 = {}, acc1 = {}, acc2 = {}, acc3 = {};
  const int kstart = chunk * 2048;
  for (int k = kstart; k < kstart + 2048; k += 32) {
    // stage B tile: lane loads K-row (k+lane), cols n0..n0+15 (fp32 -> f16)
    const float* wrow = W + (size_t)(k + lane) * 512 + n0;
    float4 w0 = *(const float4*)(wrow + 0);
    float4 w1 = *(const float4*)(wrow + 4);
    float4 w2 = *(const float4*)(wrow + 8);
    float4 w3 = *(const float4*)(wrow + 12);
    _Float16* d = lbs + lane * 16;
    d[0]  = (_Float16)w0.x; d[1]  = (_Float16)w0.y; d[2]  = (_Float16)w0.z; d[3]  = (_Float16)w0.w;
    d[4]  = (_Float16)w1.x; d[5]  = (_Float16)w1.y; d[6]  = (_Float16)w1.z; d[7]  = (_Float16)w1.w;
    d[8]  = (_Float16)w2.x; d[9]  = (_Float16)w2.y; d[10] = (_Float16)w2.z; d[11] = (_Float16)w2.w;
    d[12] = (_Float16)w3.x; d[13] = (_Float16)w3.y; d[14] = (_Float16)w3.z; d[15] = (_Float16)w3.w;
    __syncthreads();  // single-wave WG: S_NOP + ds waits
    V16H b;
#pragma unroll
    for (int j = 0; j < 16; ++j) b.e[j] = lbs[(kbB + j) * 16 + mn];
    acc0 = wmma_step(load_a_tile(feat, 98304, 0,  k, lane), b.v, acc0);
    acc1 = wmma_step(load_a_tile(feat, 98304, 16, k, lane), b.v, acc1);
    acc2 = wmma_step(load_a_tile(feat, 98304, 32, k, lane), b.v, acc2);
    acc3 = wmma_step(load_a_tile(feat, 98304, 48, k, lane), b.v, acc3);
    __syncthreads();
  }
  V8F r0, r1, r2, r3;
  r0.v = acc0; r1.v = acc1; r2.v = acc2; r3.v = acc3;
#pragma unroll
  for (int i = 0; i < 8; ++i) {
    atomicAdd(&C[(size_t)(0  + i + 8 * hi) * 512 + n0 + mn], r0.e[i]);
    atomicAdd(&C[(size_t)(16 + i + 8 * hi) * 512 + n0 + mn], r1.e[i]);
    atomicAdd(&C[(size_t)(32 + i + 8 * hi) * 512 + n0 + mn], r2.e[i]);
    atomicAdd(&C[(size_t)(48 + i + 8 * hi) * 512 + n0 + mn], r3.e[i]);
  }
}

// ---------------------------------------------------------------------------
// fc head: out(n,t,v) = h_out(t*64+n, :) @ fc_W(v,:) + fc_b.
// One wave = 64 rows x 16 cols (4 M tiles), B tile loaded once per k-step.
// grid.x = 625 (V tiles), grid.y = 32 (row blocks of 64), block = 32.
// ---------------------------------------------------------------------------
__global__ __launch_bounds__(32) void fc_gemm(
    const _Float16* __restrict__ Hout,  // (2048, 512), row = t*64+n
    const _Float16* __restrict__ Wfc,   // (10000, 512)
    const float* __restrict__ bfc,
    float* __restrict__ out) {          // (64, 32, 10000)
  const int lane = threadIdx.x;
  const int n0 = blockIdx.x * 16;
  const int m0 = blockIdx.y * 64;
  v8f acc0 = {}, acc1 = {}, acc2 = {}, acc3 = {};
  for (int k = 0; k < 512; k += 32) {
    v16h b = load_b_tile(Wfc, 512, n0, k, lane);
    acc0 = wmma_step(load_a_tile(Hout, 512, m0 + 0,  k, lane), b, acc0);
    acc1 = wmma_step(load_a_tile(Hout, 512, m0 + 16, k, lane), b, acc1);
    acc2 = wmma_step(load_a_tile(Hout, 512, m0 + 32, k, lane), b, acc2);
    acc3 = wmma_step(load_a_tile(Hout, 512, m0 + 48, k, lane), b, acc3);
  }
  const int mn = lane & 15, hi = lane >> 4;
  const float bv = bfc[n0 + mn];
  V8F r[4];
  r[0].v = acc0; r[1].v = acc1; r[2].v = acc2; r[3].v = acc3;
#pragma unroll
  for (int tm = 0; tm < 4; ++tm)
#pragma unroll
    for (int i = 0; i < 8; ++i) {
      int row = m0 + tm * 16 + i + 8 * hi;  // = t*64 + n
      int t = row >> 6, n = row & 63;
      out[((size_t)n * 32 + t) * 10000 + n0 + mn] = r[tm].e[i] + bv;
    }
}

// ---------------------------------------------------------------------------
// Attention scores + softmax + context, one block per batch row n.
// score[p] = sum_a relu(attn_h[n,a] + attn_img[n,p,a]) * wfull[a] + bfull
// ctx[n,c] = sum_p softmax(score)[p] * features[n,p,c] -> xt[n, 0..1535] (f16)
// ---------------------------------------------------------------------------
__global__ __launch_bounds__(256) void attn_ctx(
    const float* __restrict__ attn_h,   // (64, 512)
    const float* __restrict__ attn_img, // (64, 64, 512)
    const float* __restrict__ feats,    // (64, 64, 1536) fp32
    const float* __restrict__ wfull,    // (512)
    const float* __restrict__ bfull,    // (1)
    _Float16* __restrict__ xt) {        // (64, 2048), writes cols 0..1535
  const int n = blockIdx.x;
  __shared__ float score[64];
  const int wid = threadIdx.x >> 5, lane = threadIdx.x & 31;
  const float* ah = attn_h + (size_t)n * 512;
  for (int p = wid; p < 64; p += 8) {
    const float* img = attn_img + ((size_t)n * 64 + p) * 512;
    float s = 0.0f;
    for (int a = lane; a < 512; a += 32) {
      float e = ah[a] + img[a];
      e = e > 0.0f ? e : 0.0f;
      s += e * wfull[a];
    }
#pragma unroll
    for (int off = 16; off > 0; off >>= 1) s += __shfl_down(s, off, 32);
    if (lane == 0) score[p] = s + bfull[0];
  }
  __syncthreads();
  if (wid == 0) {  // softmax over 64 pixels, lanes hold 2 each
    float v0 = score[lane], v1 = score[lane + 32];
    float m = fmaxf(v0, v1);
#pragma unroll
    for (int off = 16; off > 0; off >>= 1) m = fmaxf(m, __shfl_xor(m, off, 32));
    float e0 = __expf(v0 - m), e1 = __expf(v1 - m);
    float s = e0 + e1;
#pragma unroll
    for (int off = 16; off > 0; off >>= 1) s += __shfl_xor(s, off, 32);
    float inv = 1.0f / s;
    score[lane] = e0 * inv;
    score[lane + 32] = e1 * inv;
  }
  __syncthreads();
  const float* fr = feats + (size_t)n * 64 * 1536;
  for (int c = threadIdx.x; c < 1536; c += 256) {
    float s = 0.0f;
    for (int p = 0; p < 64; ++p) s += score[p] * fr[(size_t)p * 1536 + c];
    xt[(size_t)n * 2048 + c] = (_Float16)s;
  }
}

// ---------------------------------------------------------------------------
// LSTM pointwise: c' = sig(f)*c + sig(i)*tanh(g); h' = sig(o)*tanh(c')
// ---------------------------------------------------------------------------
__global__ __launch_bounds__(256) void lstm_cell(
    const float* __restrict__ G,    // (64, 2048) gate order i,f,g,o
    float* __restrict__ c,          // (64, 512)
    _Float16* __restrict__ hnew) {  // (64, 512) f16 slot of h_out
  const int idx = blockIdx.x * blockDim.x + threadIdx.x;  // 0..32767
  const int n = idx >> 9, j = idx & 511;
  const float* g = G + (size_t)n * 2048;
  float ig = sigmoidf_(g[j]);
  float fg = sigmoidf_(g[512 + j]);
  float gg = tanhf(g[1024 + j]);
  float og = sigmoidf_(g[1536 + j]);
  float cn = fg * c[idx] + ig * gg;
  c[idx] = cn;
  hnew[idx] = (_Float16)(og * tanhf(cn));
}

// ---------------------------------------------------------------------------
// Small utilities
// ---------------------------------------------------------------------------
__global__ void cvt_f16(const float* __restrict__ in, _Float16* __restrict__ out,
                        size_t n) {
  size_t i = (size_t)blockIdx.x * blockDim.x + threadIdx.x;
  size_t stride = (size_t)gridDim.x * blockDim.x;
  for (; i < n; i += stride) out[i] = (_Float16)in[i];
}

__global__ void zero_f32(float* __restrict__ p, int n) {
  int i = blockIdx.x * blockDim.x + threadIdx.x;
  if (i < n) p[i] = 0.0f;
}

__global__ void state_init(const float* __restrict__ h0c0,
                           _Float16* __restrict__ h0_16,
                           float* __restrict__ c) {
  int idx = blockIdx.x * blockDim.x + threadIdx.x;  // 0..32767
  h0_16[idx] = (_Float16)h0c0[idx];
  c[idx] = h0c0[32768 + idx];
}

// x_buf[t][n][1536 + e] = f16(embd_W[captions[n][t]][e])
__global__ void embed_gather(const float* __restrict__ embd_W,
                             const int* __restrict__ captions,
                             _Float16* __restrict__ xbuf) {
  const int b = blockIdx.x;  // n*32 + t
  const int n = b >> 5, t = b & 31;
  const int cap = captions[n * 32 + t];
  const float* row = embd_W + (size_t)cap * 512;
  _Float16* dst = xbuf + (size_t)t * (64 * 2048) + (size_t)n * 2048 + 1536;
  for (int e = threadIdx.x; e < 512; e += blockDim.x) dst[e] = (_Float16)row[e];
}

// ---------------------------------------------------------------------------
extern "C" void kernel_launch(void* const* d_in, const int* in_sizes, int n_in,
                              void* d_out, int out_size, void* d_ws,
                              size_t ws_size, hipStream_t stream) {
  (void)in_sizes; (void)n_in; (void)out_size; (void)ws_size;
  const float* features     = (const float*)d_in[0];
  const int*   captions     = (const int*)d_in[1];
  const float* embd_W       = (const float*)d_in[2];
  const float* attn_token_W = (const float*)d_in[3];
  const float* attn_token_b = (const float*)d_in[4];
  const float* attn_feat_W  = (const float*)d_in[5];
  const float* attn_feat_b  = (const float*)d_in[6];
  const float* attn_full_W  = (const float*)d_in[7];
  const float* attn_full_b  = (const float*)d_in[8];
  const float* W_ih         = (const float*)d_in[9];
  const float* b_ih         = (const float*)d_in[10];
  const float* W_hh         = (const float*)d_in[11];
  const float* b_hh         = (const float*)d_in[12];
  const float* fc_W         = (const float*)d_in[13];
  const float* fc_b         = (const float*)d_in[14];
  const float* init_Wh      = (const float*)d_in[15];
  const float* init_Wc      = (const float*)d_in[16];
  float* out = (float*)d_out;

  // ---- workspace carve-out (aligned 256B, ~55 MB total) ----
  char* ws = (char*)d_ws;
  size_t off = 0;
  auto carve = [&](size_t bytes) -> void* {
    void* p = ws + off;
    off += (bytes + 255) & ~(size_t)255;
    return p;
  };
  _Float16* feat16   = (_Float16*)carve((size_t)64 * 64 * 1536 * 2);  // 12.6MB
  _Float16* afW16    = (_Float16*)carve((size_t)512 * 1536 * 2);
  _Float16* atW16    = (_Float16*)carve((size_t)512 * 512 * 2);
  _Float16* Wih16    = (_Float16*)carve((size_t)2048 * 2048 * 2);     // 8.4MB
  _Float16* Whh16    = (_Float16*)carve((size_t)2048 * 512 * 2);
  _Float16* fcW16    = (_Float16*)carve((size_t)10000 * 512 * 2);     // 10.2MB
  _Float16* xbuf     = (_Float16*)carve((size_t)32 * 64 * 2048 * 2);  // 8.4MB
  _Float16* hout     = (_Float16*)carve((size_t)33 * 64 * 512 * 2);   // slots 0..32
  float*    attn_img = (float*)carve((size_t)64 * 64 * 512 * 4);      // 8.4MB
  float*    attn_h   = (float*)carve((size_t)64 * 512 * 4);
  float*    gates    = (float*)carve((size_t)64 * 2048 * 4);
  float*    h0c0     = (float*)carve((size_t)2 * 64 * 512 * 4);
  float*    cbuf     = (float*)carve((size_t)64 * 512 * 4);

  auto cvt = [&](const float* src, _Float16* dst, size_t n) {
    int blocks = (int)((n + 255) / 256);
    if (blocks > 4096) blocks = 4096;
    cvt_f16<<<blocks, 256, 0, stream>>>(src, dst, n);
  };

  // ---- one-time conversions / precompute ----
  cvt(features, feat16, (size_t)64 * 64 * 1536);
  cvt(attn_feat_W, afW16, (size_t)512 * 1536);
  cvt(attn_token_W, atW16, (size_t)512 * 512);
  cvt(W_ih, Wih16, (size_t)2048 * 2048);
  cvt(W_hh, Whh16, (size_t)2048 * 512);
  cvt(fc_W, fcW16, (size_t)10000 * 512);
  zero_f32<<<256, 256, 0, stream>>>(h0c0, 2 * 64 * 512);
  embed_gather<<<2048, 256, 0, stream>>>(embd_W, captions, xbuf);

  // attn_img = features @ attn_feat_W^T + b : (4096,512), K=1536
  gemm_nk<<<dim3(32, 256), 32, 0, stream>>>(feat16, 1536, afW16, 1536,
                                            attn_feat_b, attn_img, 512, 1536);

  // h0/c0 = feat_flat @ init_Wh/Wc : split-K (48 chunks) * {h,c}
  init_state_gemm<<<dim3(32, 1, 96), 32, 0, stream>>>(feat16, init_Wh, init_Wc,
                                                      h0c0);
  state_init<<<128, 256, 0, stream>>>(h0c0, hout, cbuf);

  // ---- T=32 sequential recurrence ----
  for (int t = 0; t < 32; ++t) {
    const _Float16* hprev = hout + (size_t)t * 64 * 512;
    _Float16* xt = xbuf + (size_t)t * 64 * 2048;
    // attn_h = h @ attn_token_W^T + b : (64,512), K=512
    gemm_nk<<<dim3(32, 4), 32, 0, stream>>>(hprev, 512, atW16, 512,
                                            attn_token_b, attn_h, 512, 512);
    // scores + softmax + context -> xt[:, 0:1536]
    attn_ctx<<<64, 256, 0, stream>>>(attn_h, attn_img, features, attn_full_W,
                                     attn_full_b, xt);
    // gates = x @ W_ih^T + h @ W_hh^T + b_ih + b_hh : (64,2048)
    gates_gemm<<<dim3(128, 4), 32, 0, stream>>>(xt, Wih16, hprev, Whh16, b_ih,
                                                b_hh, gates);
    // pointwise LSTM cell; writes f16 h into h_out slot t+1
    lstm_cell<<<128, 256, 0, stream>>>(gates, cbuf,
                                       hout + (size_t)(t + 1) * 64 * 512);
  }

  // preds(n,t,v) = h_out @ fc_W^T + fc_b : (2048, 10000), K=512
  fc_gemm<<<dim3(625, 32), 32, 0, stream>>>(hout + (size_t)64 * 512, fcW16,
                                            fc_b, out);
}
